// CNFAdapter_65025804861678
// MI455X (gfx1250) — matrix operand order repaired and checked by reading, not compiled, not run on hardware
//
#include <hip/hip_runtime.h>
#include <hip/hip_bf16.h>
#include <math.h>
#include <stdint.h>

typedef __bf16 bf16;
typedef __attribute__((ext_vector_type(16))) __bf16 v16bf;
typedef __attribute__((ext_vector_type(8)))  float  v8f;

#define D_    256
#define H_    8
#define P_    32
#define HD_   32
#define NB_   32
#define NC_   2048
#define NL_   8
#define NCOMBO_ 514

static __device__ __forceinline__ v8f zero8() {
  v8f z = {0.f, 0.f, 0.f, 0.f, 0.f, 0.f, 0.f, 0.f};
  return z;
}

static __device__ __forceinline__ v8f wmma_bf16(v16bf a, v16bf b, v8f c) {
  // v_wmma_f32_16x16x32_bf16: D = A(16x32 bf16) x B(32x16 bf16) + C(16x16 f32)
  return __builtin_amdgcn_wmma_f32_16x16x32_bf16(false, a, false, b, (short)0, c,
                                                 false, false);
}

// A-fragment (16x32, 16-bit): lanes 0-15 row m=lane hold K 0..15;
// lanes 16-31 row m=lane-16 hold K 16..31 (pairs per VGPR).
static __device__ __forceinline__ v16bf load_a_frag(const bf16* A, int lda, int lane) {
  int m = lane & 15;
  int half = lane >> 4;
  const bf16* row = A + (size_t)m * lda;
  v16bf a;
#pragma unroll
  for (int p = 0; p < 8; ++p) {
    int kb = half * 16 + 2 * p;
    a[2 * p]     = row[kb];
    a[2 * p + 1] = row[kb + 1];
  }
  return a;
}

// B-fragment (32x16) where B[k][n] = W[(n0+n)*ldw + k0+k], W is f32 (converted).
static __device__ __forceinline__ v16bf load_bT_f32(const float* W, int ldw,
                                                    int n0, int k0, int lane) {
  int n = lane & 15;
  int kh = lane >> 4;
  const float* row = W + (size_t)(n0 + n) * ldw + k0 + kh * 16;
  v16bf b;
#pragma unroll
  for (int j = 0; j < 16; ++j) b[j] = (bf16)row[j];
  return b;
}

// B[k][n] = M[(r0+n)*ldm + c0k + k]  (bf16 matrix accessed transposed)
static __device__ __forceinline__ v16bf load_bT_bf16(const bf16* M, int ldm,
                                                     int r0, int c0k, int lane) {
  int n = lane & 15;
  int kh = lane >> 4;
  const bf16* row = M + (size_t)(r0 + n) * ldm + c0k + kh * 16;
  v16bf b;
#pragma unroll
  for (int j = 0; j < 16; ++j) b[j] = row[j];
  return b;
}

// B[k][n] = M[(k0row+k)*ldm + n0col + n]  (bf16 matrix, natural K-major)
static __device__ __forceinline__ v16bf load_b_colmajor(const bf16* M, int ldm,
                                                        int k0row, int n0col, int lane) {
  int n = lane & 15;
  int kh = lane >> 4;
  const bf16* base = M + (size_t)(k0row + kh * 16) * ldm + n0col + n;
  v16bf b;
#pragma unroll
  for (int j = 0; j < 8; ++j) {
    b[2 * j]     = base[(2 * j) * ldm];
    b[2 * j + 1] = base[(2 * j + 1) * ldm];
  }
  return b;
}

// ---------------------------------------------------------------------------
// Kernel 0: one-time bf16 conversion of the hot-loop weights (Wk,Wv,out_w).
// ---------------------------------------------------------------------------
__global__ __launch_bounds__(256) void wprep_kernel(
    const float* __restrict__ in_proj_w, const float* __restrict__ out_w,
    bf16* __restrict__ wkv, bf16* __restrict__ outw) {
  int i = blockIdx.x * blockDim.x + threadIdx.x;
  int stride = gridDim.x * blockDim.x;
  for (int e = i; e < 512 * 256; e += stride)
    wkv[e] = (bf16)in_proj_w[256 * 256 + e];   // rows 256..767 = Wk | Wv
  for (int e = i; e < 256 * 256; e += stride)
    outw[e] = (bf16)out_w[e];
}

// ---------------------------------------------------------------------------
// Kernel 1: literal-MLP table over the 514 distinct (var,sign) combos.
// ---------------------------------------------------------------------------
__global__ __launch_bounds__(256) void lit_table_kernel(
    const float* __restrict__ var_embed, const float* __restrict__ sign_embed,
    const float* __restrict__ W1, const float* __restrict__ b1,
    const float* __restrict__ W2, const float* __restrict__ b2,
    float* __restrict__ table) {
  __shared__ bf16 Alds[16 * 512];
  __shared__ bf16 Hlds[16 * 256];
  int tid = threadIdx.x;
  int m0 = blockIdx.x * 16;
#pragma unroll
  for (int i = 0; i < 32; ++i) {
    int e = tid + 256 * i;
    int m = e >> 9;
    int k = e & 511;
    int combo = m0 + m;
    float val = 0.f;
    if (combo < NCOMBO_) {
      int v = combo >> 1, s = combo & 1;
      val = (k < 256) ? var_embed[v * 256 + k] : sign_embed[s * 256 + (k - 256)];
    }
    Alds[m * 512 + k] = (bf16)val;
  }
  __syncthreads();
  int w = tid >> 5, lane = tid & 31;
  int n = lane & 15, half = lane >> 4, mb = half * 8;
#pragma unroll
  for (int t = 0; t < 2; ++t) {
    int nt = w + 8 * t;
    v8f acc = zero8();
    for (int k0 = 0; k0 < 512; k0 += 32) {
      v16bf a  = load_a_frag(Alds + k0, 512, lane);
      v16bf bb = load_bT_f32(W1, 512, nt * 16, k0, lane);
      acc = wmma_bf16(a, bb, acc);
    }
    float bias = b1[nt * 16 + n];
#pragma unroll
    for (int r = 0; r < 8; ++r) {
      float x = acc[r] + bias;
      float g = 0.5f * x * (1.0f + erff(x * 0.70710678f));
      Hlds[(mb + r) * 256 + nt * 16 + n] = (bf16)g;
    }
  }
  __syncthreads();
#pragma unroll
  for (int t = 0; t < 2; ++t) {
    int nt = w + 8 * t;
    v8f acc = zero8();
    for (int k0 = 0; k0 < 256; k0 += 32) {
      v16bf a  = load_a_frag(Hlds + k0, 256, lane);
      v16bf bb = load_bT_f32(W2, 256, nt * 16, k0, lane);
      acc = wmma_bf16(a, bb, acc);
    }
    float bias = b2[nt * 16 + n];
#pragma unroll
    for (int r = 0; r < 8; ++r) {
      int row = m0 + mb + r;
      if (row < NCOMBO_) table[(size_t)row * 256 + nt * 16 + n] = acc[r] + bias;
    }
  }
}

// ---------------------------------------------------------------------------
// Kernel 2: q = (prefix @ Wq^T + bq) * (1/sqrt(hd)), bf16, layout [H][P][hd]
// ---------------------------------------------------------------------------
__global__ __launch_bounds__(256) void qproj_kernel(
    const float* __restrict__ prefix, const float* __restrict__ in_proj_w,
    const float* __restrict__ in_proj_b, bf16* __restrict__ qout) {
  int tid = threadIdx.x;
  int p = tid >> 3;
  int c0 = tid & 7;
  const float scale = 0.17677669529663687f;  // 1/sqrt(32)
  for (int j = 0; j < 32; ++j) {
    int c = c0 + 8 * j;
    float acc = in_proj_b[c];
    const float* pr = prefix + p * 256;
    const float* wr = in_proj_w + (size_t)c * 256;
    for (int k = 0; k < 256; ++k) acc += pr[k] * wr[k];
    acc *= scale;
    qout[(c >> 5) * (P_ * HD_) + p * HD_ + (c & 31)] = (bf16)acc;
  }
}

// ---------------------------------------------------------------------------
// Kernel 3: clause = LN(mean_l table[idx]) ; then K/V = clause @ {Wk,Wv}^T.
// A-fragments hoisted (loaded once, reused for all 4 tiles per wave);
// weights read as bf16.
// ---------------------------------------------------------------------------
__global__ __launch_bounds__(256) void clause_kv_kernel(
    const int* __restrict__ var_idx, const int* __restrict__ sign_idx,
    const float* __restrict__ table, const float* __restrict__ cn_g,
    const float* __restrict__ cn_b, const bf16* __restrict__ wkv,
    const float* __restrict__ in_proj_b, bf16* __restrict__ kbuf,
    bf16* __restrict__ vbuf) {
  __shared__ bf16 Clds[16 * 256];
  int tid = threadIdx.x;
  int r0 = blockIdx.x * 16;
  int row = tid >> 4;
  int l16 = tid & 15;
  int gr = r0 + row;
  int off[NL_];
#pragma unroll
  for (int l = 0; l < NL_; ++l) {
    int v = var_idx[(size_t)gr * NL_ + l];
    int s = sign_idx[(size_t)gr * NL_ + l];
    off[l] = (v * 2 + s) * 256;
  }
  float x[16];
  float s1 = 0.f, s2 = 0.f;
#pragma unroll
  for (int j = 0; j < 16; ++j) {
    int d = l16 + 16 * j;
    float acc = 0.f;
#pragma unroll
    for (int l = 0; l < NL_; ++l) acc += table[off[l] + d];
    acc *= 0.125f;
    x[j] = acc;
    s1 += acc;
    s2 += acc * acc;
  }
#pragma unroll
  for (int m = 8; m >= 1; m >>= 1) {
    s1 += __shfl_xor(s1, m, 16);
    s2 += __shfl_xor(s2, m, 16);
  }
  float mu  = s1 * (1.0f / 256.0f);
  float var = s2 * (1.0f / 256.0f) - mu * mu;
  float inv = rsqrtf(var + 1e-5f);
#pragma unroll
  for (int j = 0; j < 16; ++j) {
    int d = l16 + 16 * j;
    Clds[row * 256 + d] = (bf16)((x[j] - mu) * inv * cn_g[d] + cn_b[d]);
  }
  __syncthreads();
  int w = tid >> 5, lane = tid & 31;
  int n = lane & 15, half = lane >> 4, mb = half * 8;
  // hoist the 8 shared A-fragments once per wave
  v16bf afr[8];
#pragma unroll
  for (int kk = 0; kk < 8; ++kk)
    afr[kk] = load_a_frag(Clds + kk * 32, 256, lane);
#pragma unroll
  for (int t = 0; t < 4; ++t) {
    int tile = w * 4 + t;
    int mat = tile >> 4;  // 0 = K, 1 = V
    int nt = tile & 15;
    const bf16* Wm = wkv + (size_t)mat * 256 * 256;
    v8f acc = zero8();
#pragma unroll
    for (int kk = 0; kk < 8; ++kk) {
      v16bf bb = load_bT_bf16(Wm, 256, nt * 16, kk * 32, lane);
      acc = wmma_bf16(afr[kk], bb, acc);
    }
    float bias = in_proj_b[256 + mat * 256 + nt * 16 + n];
    bf16* dst = mat ? vbuf : kbuf;
#pragma unroll
    for (int r = 0; r < 8; ++r)
      dst[(size_t)(r0 + mb + r) * 256 + nt * 16 + n] = (bf16)(acc[r] + bias);
  }
}

// ---------------------------------------------------------------------------
// Kernel 4: flash attention, one block per (b,h). Each wave owns 8 disjoint
// 32-column chunks of C with online softmax; WMMA for scores and ctx.
// V sub-tiles staged to LDS with global_load_async_to_lds_b128 (ASYNCcnt).
// ---------------------------------------------------------------------------
__global__ __launch_bounds__(256) void attn_kernel(
    const bf16* __restrict__ qbuf, const bf16* __restrict__ kbuf,
    const bf16* __restrict__ vbuf, const int* __restrict__ mask,
    float* __restrict__ ctx) {
  __shared__ bf16 probL[8][16 * 32];
  __shared__ bf16 Vt[8][32 * 32];      // per-wave async-staged V sub-tile
  __shared__ float ctxP[8][32][32];
  __shared__ float stM[8][32];
  __shared__ float stL[8][32];
  int tid = threadIdx.x;
  int w = tid >> 5, lane = tid & 31;
  int b = blockIdx.x >> 3, h = blockIdx.x & 7;
  int n = lane & 15, half = lane >> 4;

  v16bf qa[2];
#pragma unroll
  for (int mt = 0; mt < 2; ++mt)
    qa[mt] = load_a_frag(qbuf + h * (P_ * HD_) + mt * 16 * HD_, HD_, lane);

  v8f acc[2][2];
  float rowm[2][8], rowl[2][8];
#pragma unroll
  for (int mt = 0; mt < 2; ++mt)
#pragma unroll
    for (int r = 0; r < 8; ++r) { rowm[mt][r] = -1e30f; rowl[mt][r] = 0.f; }
#pragma unroll
  for (int mt = 0; mt < 2; ++mt)
#pragma unroll
    for (int et = 0; et < 2; ++et) acc[mt][et] = zero8();

  const bf16* kptr = kbuf + h * 32;
  const bf16* vptr = vbuf + h * 32;
  // lane's async-copy task: 32 rows x 64B, 4 b128 segments per row
  int arow = lane >> 2;       // +8 per i
  int aseg = lane & 3;

  for (int cc = w; cc < 64; cc += 8) {
    int c0 = cc * 32;
    if (c0 + 256 < NC_)
      __builtin_prefetch(kptr + (size_t)(c0 + 256) * 256, 0, 1);
    // make sure previous chunk's LDS reads of Vt[w] have completed
    asm volatile("s_wait_dscnt 0x0" ::: "memory");
    // async-stage the 32x32 bf16 V sub-tile for this chunk into LDS
#pragma unroll
    for (int i = 0; i < 4; ++i) {
      int rrow = arow + 8 * i;
      unsigned ldsoff = (unsigned)(uintptr_t)&Vt[w][rrow * 32 + aseg * 8];
      uint64_t ga = (uint64_t)(uintptr_t)(vptr + (size_t)(c0 + rrow) * 256 + aseg * 8);
      asm volatile("global_load_async_to_lds_b128 %0, %1, off"
                   :: "v"(ldsoff), "v"(ga) : "memory");
    }
    v16bf bk[2];
#pragma unroll
    for (int ct = 0; ct < 2; ++ct)
      bk[ct] = load_bT_bf16(kptr, 256, c0 + ct * 16, 0, lane);  // B[e][c]=K[c][e]
    int mk0 = mask[(size_t)b * NC_ + c0 + n];
    int mk1 = mask[(size_t)b * NC_ + c0 + 16 + n];
    // wait for the async V tile, then build B-fragments from LDS
    asm volatile("s_wait_asynccnt 0x0" ::: "memory");
    v16bf bv[2];
#pragma unroll
    for (int et = 0; et < 2; ++et)
      bv[et] = load_b_colmajor(&Vt[w][0], 32, 0, et * 16, lane);

#pragma unroll
    for (int mt = 0; mt < 2; ++mt) {
      v8f s0 = wmma_bf16(qa[mt], bk[0], zero8());
      v8f s1 = wmma_bf16(qa[mt], bk[1], zero8());
#pragma unroll
      for (int r = 0; r < 8; ++r) {
        if (mk0 <= 0) s0[r] = -1e9f;
        if (mk1 <= 0) s1[r] = -1e9f;
      }
#pragma unroll
      for (int r = 0; r < 8; ++r) {
        float cm = fmaxf(s0[r], s1[r]);
#pragma unroll
        for (int m = 8; m >= 1; m >>= 1) cm = fmaxf(cm, __shfl_xor(cm, m, 16));
        float mnew = fmaxf(rowm[mt][r], cm);
        float sc = __expf(rowm[mt][r] - mnew);
        rowm[mt][r] = mnew;
        float p0 = __expf(s0[r] - mnew);
        float p1 = __expf(s1[r] - mnew);
        float ps = p0 + p1;
#pragma unroll
        for (int m = 8; m >= 1; m >>= 1) ps += __shfl_xor(ps, m, 16);
        rowl[mt][r] = rowl[mt][r] * sc + ps;
#pragma unroll
        for (int et = 0; et < 2; ++et) acc[mt][et][r] *= sc;
        probL[w][(r + 8 * half) * 32 + n]      = (bf16)p0;
        probL[w][(r + 8 * half) * 32 + 16 + n] = (bf16)p1;
      }
      v16bf pa = load_a_frag(&probL[w][0], 32, lane);
#pragma unroll
      for (int et = 0; et < 2; ++et) acc[mt][et] = wmma_bf16(pa, bv[et], acc[mt][et]);
    }
  }
  // publish per-wave partials
#pragma unroll
  for (int mt = 0; mt < 2; ++mt)
#pragma unroll
    for (int r = 0; r < 8; ++r) {
      int m = mt * 16 + r + 8 * half;
#pragma unroll
      for (int et = 0; et < 2; ++et) ctxP[w][m][et * 16 + n] = acc[mt][et][r];
      if (n == 0) { stM[w][m] = rowm[mt][r]; stL[w][m] = rowl[mt][r]; }
    }
  __syncthreads();
  // combine across waves
  int p = tid >> 3;
  int e0 = (tid & 7) * 4;
  float gm = -1e30f;
#pragma unroll
  for (int ww = 0; ww < 8; ++ww) gm = fmaxf(gm, stM[ww][p]);
  float den = 0.f, wgt[8];
#pragma unroll
  for (int ww = 0; ww < 8; ++ww) {
    wgt[ww] = __expf(stM[ww][p] - gm);
    den += stL[ww][p] * wgt[ww];
  }
  float rden = 1.0f / den;
#pragma unroll
  for (int e = e0; e < e0 + 4; ++e) {
    float num = 0.f;
#pragma unroll
    for (int ww = 0; ww < 8; ++ww) num += ctxP[ww][p][e] * wgt[ww];
    ctx[(size_t)(b * P_ + p) * 256 + h * 32 + e] = num * rden;
  }
}

// ---------------------------------------------------------------------------
// Kernel 5: out = LN(prefix + ctx @ out_w^T + out_b)
// ---------------------------------------------------------------------------
__global__ __launch_bounds__(256) void outproj_kernel(
    const float* __restrict__ ctx, const bf16* __restrict__ outw,
    const float* __restrict__ out_b, const float* __restrict__ prefix,
    const float* __restrict__ pn_g, const float* __restrict__ pn_b,
    float* __restrict__ out) {
  __shared__ bf16 Clds[16 * 256];
  __shared__ float Rlds[16 * 256];
  int tid = threadIdx.x;
  int r0 = blockIdx.x * 16;
#pragma unroll
  for (int i = 0; i < 16; ++i) {
    int e = tid + 256 * i;
    Clds[e] = (bf16)ctx[(size_t)r0 * 256 + e];
  }
  __syncthreads();
  int w = tid >> 5, lane = tid & 31;
  int n = lane & 15, half = lane >> 4, mb = half * 8;
  v16bf afr[8];
#pragma unroll
  for (int kk = 0; kk < 8; ++kk)
    afr[kk] = load_a_frag(Clds + kk * 32, 256, lane);
#pragma unroll
  for (int t = 0; t < 2; ++t) {
    int nt = w + 8 * t;
    v8f acc = zero8();
#pragma unroll
    for (int kk = 0; kk < 8; ++kk) {
      v16bf bb = load_bT_bf16(outw, 256, nt * 16, kk * 32, lane);
      acc = wmma_bf16(afr[kk], bb, acc);
    }
    float bias = out_b[nt * 16 + n];
#pragma unroll
    for (int r = 0; r < 8; ++r) {
      int grow = r0 + mb + r;
      int p = grow & 31;
      Rlds[(mb + r) * 256 + nt * 16 + n] =
          acc[r] + bias + prefix[p * 256 + nt * 16 + n];
    }
  }
  __syncthreads();
  int row = tid >> 4, l16 = tid & 15;
  float x[16], s1 = 0.f, s2 = 0.f;
#pragma unroll
  for (int j = 0; j < 16; ++j) {
    float v = Rlds[row * 256 + l16 + 16 * j];
    x[j] = v; s1 += v; s2 += v * v;
  }
#pragma unroll
  for (int m = 8; m >= 1; m >>= 1) {
    s1 += __shfl_xor(s1, m, 16);
    s2 += __shfl_xor(s2, m, 16);
  }
  float mu  = s1 * (1.f / 256.f);
  float var = s2 * (1.f / 256.f) - mu * mu;
  float inv = rsqrtf(var + 1e-5f);
  int gr = r0 + row;
#pragma unroll
  for (int j = 0; j < 16; ++j) {
    int d = l16 + 16 * j;
    out[(size_t)gr * 256 + d] = (x[j] - mu) * inv * pn_g[d] + pn_b[d];
  }
}

extern "C" void kernel_launch(void* const* d_in, const int* in_sizes, int n_in,
                              void* d_out, int out_size, void* d_ws, size_t ws_size,
                              hipStream_t stream) {
  const int*   var_idx    = (const int*)d_in[0];
  const int*   sign_idx   = (const int*)d_in[1];
  const int*   mask       = (const int*)d_in[2];
  const float* var_embed  = (const float*)d_in[3];
  const float* sign_embed = (const float*)d_in[4];
  const float* W1         = (const float*)d_in[5];
  const float* b1         = (const float*)d_in[6];
  const float* W2         = (const float*)d_in[7];
  const float* b2         = (const float*)d_in[8];
  const float* cn_g       = (const float*)d_in[9];
  const float* cn_b       = (const float*)d_in[10];
  const float* prefix     = (const float*)d_in[11];
  const float* in_proj_w  = (const float*)d_in[12];
  const float* in_proj_b  = (const float*)d_in[13];
  const float* out_w      = (const float*)d_in[14];
  const float* out_b      = (const float*)d_in[15];
  const float* pn_g       = (const float*)d_in[16];
  const float* pn_b       = (const float*)d_in[17];

  char* ws = (char*)d_ws;
  float* table = (float*)(ws + 0);          // 514*256*4  = 526,336 B
  bf16*  qbuf  = (bf16*)(ws + 528384);      // 16,384 B
  bf16*  kbuf  = (bf16*)(ws + 544768);      // 33,554,432 B
  bf16*  vbuf  = (bf16*)(ws + 34099200);    // 33,554,432 B
  float* ctxb  = (float*)(ws + 67653632);   // 1,048,576 B
  bf16*  wkv   = (bf16*)(ws + 68702208);    // 512*256*2 = 262,144 B
  bf16*  outw  = (bf16*)(ws + 68964352);    // 256*256*2 = 131,072 B

  wprep_kernel<<<256, 256, 0, stream>>>(in_proj_w, out_w, wkv, outw);
  lit_table_kernel<<<33, 256, 0, stream>>>(var_embed, sign_embed, W1, b1, W2, b2, table);
  qproj_kernel<<<1, 256, 0, stream>>>(prefix, in_proj_w, in_proj_b, qbuf);
  clause_kv_kernel<<<4096, 256, 0, stream>>>(var_idx, sign_idx, table, cn_g, cn_b,
                                             wkv, in_proj_b, kbuf, vbuf);
  attn_kernel<<<NB_ * H_, 256, 0, stream>>>(qbuf, kbuf, vbuf, mask, ctxb);
  outproj_kernel<<<(NB_ * P_) / 16, 256, 0, stream>>>(ctxb, outw, out_b, prefix,
                                                      pn_g, pn_b, (float*)d_out);
}
